// MaskedMHCA_15728170238324
// MI455X (gfx1250) — matrix operand order repaired and checked by reading, not compiled, not run on hardware
//
#include <hip/hip_runtime.h>

typedef __attribute__((ext_vector_type(16))) _Float16 v16h;
typedef __attribute__((ext_vector_type(8)))  _Float16 v8h;
typedef __attribute__((ext_vector_type(8)))  float    v8f;

#define B_   2
#define C_   1024
#define T_   2048
#define NH_  16
#define CH_  64
#define EPS_ 1e-5f
#define SQ2_ 0.70710678118654752f

static __device__ __forceinline__ v8f vzero8() {
  v8f r;
#pragma unroll
  for (int i = 0; i < 8; ++i) r[i] = 0.0f;
  return r;
}

static __device__ __forceinline__ v16h join8(v8h lo, v8h hi) {
  v16h r;
#pragma unroll
  for (int i = 0; i < 8; ++i) { r[i] = lo[i]; r[i + 8] = hi[i]; }
  return r;
}

// Async copy of 16 bytes global -> LDS (per-lane addresses), ASYNCcnt-tracked.
static __device__ __forceinline__ void async_g2l_b128(const void* gptr, void* lptr) {
  uint64_t ga = (uint64_t)(uintptr_t)gptr;
  uint32_t la = (uint32_t)(uintptr_t)lptr;  // LDS aperture: low 32 bits = LDS byte address
  asm volatile("global_load_async_to_lds_b128 %0, %1, off" ::"v"(la), "v"(ga) : "memory");
}
static __device__ __forceinline__ void wait_async0() {
  asm volatile("s_wait_asynccnt 0x0" ::: "memory");
}

// ---------------------------------------------------------------------------
// Weight conversion f32 -> f16
// ---------------------------------------------------------------------------
__global__ void cvt16_kernel(const float* __restrict__ s, _Float16* __restrict__ d, int n) {
  int i = blockIdx.x * blockDim.x + threadIdx.x;
  if (i < n) d[i] = (_Float16)s[i];
}

// mask (bool bytes) -> float tail of d_out (tuple second element)
__global__ void maskout_kernel(const unsigned char* __restrict__ m, float* __restrict__ d, int n) {
  int i = blockIdx.x * blockDim.x + threadIdx.x;
  if (i < n) d[i] = m[i] ? 1.0f : 0.0f;
}

// ---------------------------------------------------------------------------
// DWT (Haar even/odd along T) + per-column LayerNorm over C, three sets of
// affine params, fp16 outputs laid out [b][c][t] (GEMM B-operand layout).
// One 256-thread block per (b, t_out) column; 4 channels per thread.
// ---------------------------------------------------------------------------
__global__ __launch_bounds__(256) void dwtln_kernel(
    const float* __restrict__ x, const unsigned char* __restrict__ mask,
    const float* __restrict__ qw, const float* __restrict__ qb,
    const float* __restrict__ kw, const float* __restrict__ kb,
    const float* __restrict__ vw, const float* __restrict__ vb,
    _Float16* __restrict__ Xq, _Float16* __restrict__ Xk, _Float16* __restrict__ Xv) {
  __shared__ float red[256];
  const int t = blockIdx.x;
  const int b = blockIdx.y;
  const int tid = threadIdx.x;
  const int half = T_ / 2;
  const float mk = mask[(size_t)b * T_ + t] ? 1.0f : 0.0f;

  float vals[4];
  float s = 0.0f;
#pragma unroll
  for (int i = 0; i < 4; ++i) {
    int c = tid + i * 256;
    const float* xc = x + ((size_t)b * C_ + c) * T_;
    float v;
    if (t < half) {
      v = SQ2_ * (xc[2 * t] + xc[2 * t + 1]);
    } else {
      int th = t - half;
      v = SQ2_ * (xc[2 * th] - xc[2 * th + 1]);
    }
    v *= mk;
    vals[i] = v;
    s += v;
  }
  red[tid] = s;
  __syncthreads();
  for (int off = 128; off > 0; off >>= 1) {
    if (tid < off) red[tid] += red[tid + off];
    __syncthreads();
  }
  const float mu = red[0] * (1.0f / C_);
  __syncthreads();

  float s2 = 0.0f;
#pragma unroll
  for (int i = 0; i < 4; ++i) { float r = vals[i] - mu; s2 += r * r; }
  red[tid] = s2;
  __syncthreads();
  for (int off = 128; off > 0; off >>= 1) {
    if (tid < off) red[tid] += red[tid + off];
    __syncthreads();
  }
  const float rstd = rsqrtf(red[0] * (1.0f / C_) + EPS_);

#pragma unroll
  for (int i = 0; i < 4; ++i) {
    int c = tid + i * 256;
    float n = (vals[i] - mu) * rstd;
    size_t o = ((size_t)b * C_ + c) * T_ + t;
    Xq[o] = (_Float16)(n * qw[c] + qb[c]);
    Xk[o] = (_Float16)(n * kw[c] + kb[c]);
    Xv[o] = (_Float16)(n * vw[c] + vb[c]);
  }
}

// ---------------------------------------------------------------------------
// WMMA GEMM: out[b][o][t] = sum_c W[o][c] * X[b][c][t] + bias[o]
// BM=128, BN=128, BK=32; 256 threads = 8 waves (2x4), each wave 64x32 = 4x2
// 16x16 fragments. Double-buffered LDS; A-tile prefetched with
// global_load_async_to_lds_b128 (ASYNCcnt), B-tile global loads issued before
// the WMMA block and LDS-transposed after it, so both streams overlap math.
// mode 0: fp16 store to [b][head][t][ch]  (q/k/v for attention)
// mode 1: f32 store to d_out [b][o][t], multiplied by mask[b][t]
// ---------------------------------------------------------------------------
__global__ __launch_bounds__(256) void gemm_wmma(
    const _Float16* __restrict__ W,       // C_ x C_, row-major [o][c]
    const _Float16* __restrict__ X,       // [b][c][t]
    const float* __restrict__ bias,       // C_
    const unsigned char* __restrict__ mask,  // [b][t] (mode 1)
    _Float16* __restrict__ out16,         // mode 0
    float* __restrict__ out32,            // mode 1
    int mode) {
  constexpr int BM = 128, BN = 128, BK = 32;
  constexpr int NSTEP = C_ / BK;
  __shared__ alignas(32) _Float16 As[2][BM][BK];  // [buf][row m][k]
  __shared__ alignas(32) _Float16 Bs[2][BN][BK];  // [buf][col n][k] (transposed)

  const int tid = threadIdx.x;
  const int lane = tid & 31;
  const int wid = tid >> 5;
  const int g = lane >> 4;
  const int l15 = lane & 15;
  const int wm = wid >> 2;  // 0..1
  const int wn = wid & 3;   // 0..3
  const int m0 = blockIdx.y * BM;
  const int n0 = blockIdx.x * BN;
  const int b = blockIdx.z;
  const _Float16* __restrict__ Xb = X + (size_t)b * C_ * T_;

  // per-thread tile-load coordinates
  const int ar0 = (tid * 2) >> 3;          // unused helper (kept simple below)
  (void)ar0;

  v8f acc[4][2];
#pragma unroll
  for (int mt = 0; mt < 4; ++mt)
#pragma unroll
    for (int nt = 0; nt < 2; ++nt) acc[mt][nt] = vzero8();

  v8h breg[2];

  auto issueA = [&](int kk, int buf) {
#pragma unroll
    for (int i = 0; i < 2; ++i) {
      int idx = tid + i * 256;          // 0..511
      int r = idx >> 2, vc = idx & 3;   // 128 rows x 4 vectors
      async_g2l_b128(&W[(size_t)(m0 + r) * C_ + kk + vc * 8], &As[buf][r][vc * 8]);
    }
  };
  auto fetchB = [&](int kk) {
#pragma unroll
    for (int i = 0; i < 2; ++i) {
      int idx = tid + i * 256;
      int c = idx >> 4, tv = idx & 15;  // 32 k-rows x 16 vectors
      breg[i] = *(const v8h*)&Xb[(size_t)(kk + c) * T_ + n0 + tv * 8];
    }
  };
  auto storeB = [&](int buf) {
#pragma unroll
    for (int i = 0; i < 2; ++i) {
      int idx = tid + i * 256;
      int c = idx >> 4, tv = idx & 15;
#pragma unroll
      for (int j = 0; j < 8; ++j) Bs[buf][tv * 8 + j][c] = breg[i][j];
    }
  };

  // Prologue: fill buffer 0.
  issueA(0, 0);
  fetchB(0);
  storeB(0);
  wait_async0();
  __syncthreads();

  for (int step = 0; step < NSTEP; ++step) {
    const int buf = step & 1;
    if (step + 1 < NSTEP) {
      issueA((step + 1) * BK, buf ^ 1);  // async, no wait yet
      fetchB((step + 1) * BK);           // global loads in flight during WMMAs
    }

    // A fragments: m = l15, k = 16g + h (contiguous in As[m][k])
    v16h af[4];
#pragma unroll
    for (int mt = 0; mt < 4; ++mt) {
      int r = wm * 64 + mt * 16 + l15;
      af[mt] = *(const v16h*)&As[buf][r][16 * g];
    }
    // B fragments: n = l15, k = 16g + h (contiguous in Bs[n][k])
    v16h bf[2];
#pragma unroll
    for (int nt = 0; nt < 2; ++nt) {
      int cidx = wn * 32 + nt * 16 + l15;
      bf[nt] = *(const v16h*)&Bs[buf][cidx][16 * g];
    }
#pragma unroll
    for (int mt = 0; mt < 4; ++mt)
#pragma unroll
      for (int nt = 0; nt < 2; ++nt)
        acc[mt][nt] = __builtin_amdgcn_wmma_f32_16x16x32_f16(
            false, af[mt], false, bf[nt], (short)0, acc[mt][nt], false, false);

    if (step + 1 < NSTEP) storeB(buf ^ 1);  // transpose into next buffer
    wait_async0();
    __syncthreads();
  }

  // D layout: row = v + 8g, col = l15
#pragma unroll
  for (int mt = 0; mt < 4; ++mt)
#pragma unroll
    for (int nt = 0; nt < 2; ++nt)
#pragma unroll
      for (int v = 0; v < 8; ++v) {
        int row = m0 + wm * 64 + mt * 16 + v + 8 * g;
        int col = n0 + wn * 32 + nt * 16 + l15;
        float val = acc[mt][nt][v] + bias[row];
        if (mode == 0) {
          int hd = row >> 6, ch = row & 63;
          out16[(((size_t)b * NH_ + hd) * T_ + col) * CH_ + ch] = (_Float16)val;
        } else {
          float mk = mask[(size_t)b * T_ + col] ? 1.0f : 0.0f;
          out32[((size_t)b * C_ + row) * T_ + col] = val * mk;
        }
      }
}

// ---------------------------------------------------------------------------
// Flash attention: one block per (b, h, 64 query rows); 4 waves x 16 rows.
// Online softmax in f32; V pre-multiplied by mask (reference semantics:
// softmax is unmasked, V is masked). K tile arrives via async-to-LDS copy,
// overlapping the masked/transposed V tile load. Output fp16 [b][h*64+c][t]
// so the final projection consumes it directly as a GEMM B operand.
// ---------------------------------------------------------------------------
__global__ __launch_bounds__(128) void attn_wmma(
    const _Float16* __restrict__ Q,  // [b][h][t][c]
    const _Float16* __restrict__ K,  // [b][h][t][c]
    const _Float16* __restrict__ V,  // [b][h][t][c]
    const unsigned char* __restrict__ mask,  // [b][t]
    _Float16* __restrict__ AO) {     // [b][h*64+c][t]
  __shared__ alignas(32) _Float16 Ks[32][64];      // [key][c]
  __shared__ alignas(32) _Float16 Vs[64][32];      // [c][key] (transposed)
  __shared__ alignas(32) _Float16 Ps[4][16][32];   // per-wave P staging

  const int tid = threadIdx.x;
  const int lane = tid & 31;
  const int wid = tid >> 5;
  const int g = lane >> 4;
  const int l15 = lane & 15;
  const int b = blockIdx.z;
  const int h = blockIdx.y;
  const int q0 = blockIdx.x * 64 + wid * 16;
  const float scale = 0.125f;  // 1/sqrt(64)
  const size_t headoff = ((size_t)b * NH_ + h) * (size_t)T_ * CH_;

  // Q fragments (A layout): m = l15 -> global row q0+l15; two frags cover c 0..63
  v16h qf[2];
  {
    const _Float16* __restrict__ qrow = Q + headoff + (size_t)(q0 + l15) * CH_;
#pragma unroll
    for (int f = 0; f < 2; ++f) {
      qf[f] = *(const v16h*)&qrow[32 * f + 16 * g];
    }
  }

  float m_r[8], l_r[8];
  v8f O[4];
#pragma unroll
  for (int v = 0; v < 8; ++v) { m_r[v] = -1e30f; l_r[v] = 0.0f; }
#pragma unroll
  for (int ct = 0; ct < 4; ++ct) O[ct] = vzero8();

  for (int ks = 0; ks < T_; ks += 32) {
    __syncthreads();
    // K tile: 32 keys x 64 ch — async global->LDS copy
#pragma unroll
    for (int i = 0; i < 2; ++i) {
      int idx = tid + i * 128;
      int key = idx >> 3, cv = idx & 7;
      async_g2l_b128(&K[headoff + (size_t)(ks + key) * CH_ + cv * 8], &Ks[key][cv * 8]);
    }
    // V tile, transposed, mask folded in (overlaps the async K copy)
#pragma unroll
    for (int i = 0; i < 2; ++i) {
      int idx = tid + i * 128;
      int key = idx >> 3, cv = idx & 7;
      float mk = mask[(size_t)b * T_ + ks + key] ? 1.0f : 0.0f;
      v8h tmp = *(const v8h*)&V[headoff + (size_t)(ks + key) * CH_ + cv * 8];
#pragma unroll
      for (int j = 0; j < 8; ++j) Vs[cv * 8 + j][key] = (_Float16)((float)tmp[j] * mk);
    }
    wait_async0();
    __syncthreads();

    // S = scale * Q K^T for 2 x 16-key fragments (k-dim = channels, 2 steps)
    v8f S[2];
#pragma unroll
    for (int jh = 0; jh < 2; ++jh) {
      S[jh] = vzero8();
#pragma unroll
      for (int cf = 0; cf < 2; ++cf) {
        v16h kf = *(const v16h*)&Ks[jh * 16 + l15][32 * cf + 16 * g];
        S[jh] = __builtin_amdgcn_wmma_f32_16x16x32_f16(
            false, qf[cf], false, kf, (short)0, S[jh], false, false);
      }
#pragma unroll
      for (int i = 0; i < 8; ++i) S[jh][i] *= scale;
    }

    // Online softmax; row v+8g cols live in the 16 lanes of this half-wave
#pragma unroll
    for (int v = 0; v < 8; ++v) {
      float s0 = S[0][v], s1 = S[1][v];
      float mx = fmaxf(s0, s1);
#pragma unroll
      for (int off = 1; off < 16; off <<= 1) mx = fmaxf(mx, __shfl_xor(mx, off, 32));
      float mnew = fmaxf(m_r[v], mx);
      float corr = __expf(m_r[v] - mnew);
      float p0 = __expf(s0 - mnew);
      float p1 = __expf(s1 - mnew);
      float rs = p0 + p1;
#pragma unroll
      for (int off = 1; off < 16; off <<= 1) rs += __shfl_xor(rs, off, 32);
      l_r[v] = l_r[v] * corr + rs;
      m_r[v] = mnew;
#pragma unroll
      for (int ct = 0; ct < 4; ++ct) O[ct][v] *= corr;
      int r = v + 8 * g;
      Ps[wid][r][l15]      = (_Float16)p0;
      Ps[wid][r][16 + l15] = (_Float16)p1;
    }
    __syncthreads();

    // P (A layout, k = 32 keys) x V (B layout from transposed Vs)
    v16h pf;
    {
      pf = *(const v16h*)&Ps[wid][l15][16 * g];
    }
#pragma unroll
    for (int ct = 0; ct < 4; ++ct) {
      v16h vf = *(const v16h*)&Vs[ct * 16 + l15][16 * g];
      O[ct] = __builtin_amdgcn_wmma_f32_16x16x32_f16(
          false, pf, false, vf, (short)0, O[ct], false, false);
    }
  }

  // Normalize and store: row = q0 + v + 8g, ch = ct*16 + l15
#pragma unroll
  for (int ct = 0; ct < 4; ++ct)
#pragma unroll
    for (int v = 0; v < 8; ++v) {
      int qrow = q0 + v + 8 * g;
      int c = ct * 16 + l15;
      float val = O[ct][v] / fmaxf(l_r[v], 1e-20f);
      AO[((size_t)b * C_ + h * CH_ + c) * T_ + qrow] = (_Float16)val;
    }
}

// ---------------------------------------------------------------------------
// Host-side orchestration
// ---------------------------------------------------------------------------
extern "C" void kernel_launch(void* const* d_in, const int* in_sizes, int n_in,
                              void* d_out, int out_size, void* d_ws, size_t ws_size,
                              hipStream_t stream) {
  const float* x  = (const float*)d_in[0];
  const unsigned char* mask = (const unsigned char*)d_in[1];
  const float* qw = (const float*)d_in[2];
  const float* qb = (const float*)d_in[3];
  const float* kw = (const float*)d_in[4];
  const float* kb = (const float*)d_in[5];
  const float* vw = (const float*)d_in[6];
  const float* vb = (const float*)d_in[7];
  const float* Wq = (const float*)d_in[8];
  const float* bq = (const float*)d_in[9];
  const float* Wk = (const float*)d_in[10];
  const float* bk = (const float*)d_in[11];
  const float* Wv = (const float*)d_in[12];
  const float* bv = (const float*)d_in[13];
  const float* Wp = (const float*)d_in[14];
  const float* bp = (const float*)d_in[15];

  char* ws = (char*)d_ws;
  const size_t CC2  = (size_t)C_ * C_ * sizeof(_Float16);        // 2 MB
  const size_t BCT2 = (size_t)B_ * C_ * T_ * sizeof(_Float16);   // 8 MB
  _Float16* Wq16 = (_Float16*)(ws + 0 * CC2);
  _Float16* Wk16 = (_Float16*)(ws + 1 * CC2);
  _Float16* Wv16 = (_Float16*)(ws + 2 * CC2);
  _Float16* Wp16 = (_Float16*)(ws + 3 * CC2);
  char* p = ws + 4 * CC2;
  _Float16* Xq = (_Float16*)(p + 0 * BCT2);
  _Float16* Xk = (_Float16*)(p + 1 * BCT2);
  _Float16* Xv = (_Float16*)(p + 2 * BCT2);
  _Float16* Qh = (_Float16*)(p + 3 * BCT2);
  _Float16* Kh = (_Float16*)(p + 4 * BCT2);
  _Float16* Vh = (_Float16*)(p + 5 * BCT2);
  _Float16* AO = (_Float16*)(p + 6 * BCT2);

  const int ncc = C_ * C_;
  cvt16_kernel<<<(ncc + 255) / 256, 256, 0, stream>>>(Wq, Wq16, ncc);
  cvt16_kernel<<<(ncc + 255) / 256, 256, 0, stream>>>(Wk, Wk16, ncc);
  cvt16_kernel<<<(ncc + 255) / 256, 256, 0, stream>>>(Wv, Wv16, ncc);
  cvt16_kernel<<<(ncc + 255) / 256, 256, 0, stream>>>(Wp, Wp16, ncc);

  dwtln_kernel<<<dim3(T_, B_), 256, 0, stream>>>(x, mask, qw, qb, kw, kb, vw, vb,
                                                 Xq, Xk, Xv);

  dim3 gg(T_ / 128, C_ / 128, B_);
  gemm_wmma<<<gg, 256, 0, stream>>>(Wq16, Xq, bq, mask, Qh, nullptr, 0);
  gemm_wmma<<<gg, 256, 0, stream>>>(Wk16, Xk, bk, mask, Kh, nullptr, 0);
  gemm_wmma<<<gg, 256, 0, stream>>>(Wv16, Xv, bv, mask, Vh, nullptr, 0);

  attn_wmma<<<dim3(T_ / 64, NH_, B_), 128, 0, stream>>>(Qh, Kh, Vh, mask, AO);

  gemm_wmma<<<gg, 256, 0, stream>>>(Wp16, AO, bp, mask, nullptr, (float*)d_out, 1);

  // Tuple output: second element is the (pass-through) mask.
  const size_t main_elems = (size_t)B_ * C_ * T_;
  const int nm = B_ * T_;
  if ((size_t)out_size >= main_elems + (size_t)nm) {
    maskout_kernel<<<(nm + 255) / 256, 256, 0, stream>>>(
        mask, (float*)d_out + main_elems, nm);
  }
}
